// DecoderModule_6957847020140
// MI455X (gfx1250) — matrix operand order repaired and checked by reading, compiled, not run on hardware
//
#include <hip/hip_runtime.h>

typedef _Float16 h16 __attribute__((ext_vector_type(16)));
typedef _Float16 h8  __attribute__((ext_vector_type(8)));
typedef float    f8  __attribute__((ext_vector_type(8)));

#define NODES 128
#define DIM   128
#define MHID  256
#define DOUT  16
#define EPSV  1e-12f

// ---------------- LDS layout (bytes) ----------------
#define OFF_HH      0          // 128x128 half (row major H)        32768
#define OFF_HHT     32768      // 128x128 half (H transposed)       32768
#define OFF_AGG     65536      // 128x128 half (Adj^T @ H)          32768
#define OFF_WS1T    98304      // Ws1 transposed [n][k] half        32768
#define OFF_WN1T    131072
#define OFF_WS2T    163840
#define OFF_WN2T    196608
#define OFF_MW1HT   229376     // mW1[128:256,:] transposed [256][128] half  65536
#define OFF_ADJB    294912     // 128 x 4 u32 bitsets of T rows      2048
#define OFF_ANCB    296960     // ancestor bitsets                   2048
#define OFF_CARR    299008     // 128x128 u8 suppression counts     16384
#define OFF_U       315392     // 256 f32
#define OFF_VZ      316416     // 256 f32 (mb1 + z-slab)
#define OFF_W2      317440     // 256 f32
#define OFF_HT      318464     // 128 f32  h_t (post GNN)
#define OFF_PET     318976     // 128 f32  pos_emb[t]
#define OFF_LOGIT   319488     // 128 f32
#define OFF_B1      320000     // 128 f32
#define OFF_B2      320512     // 128 f32
#define OFF_RED     321024     // red[0..7] + tot at red[8]
#define SMEM_BYTES  321280

__device__ __forceinline__ f8 wmma16(h16 a, h16 b, f8 c) {
  // v_wmma_f32_16x16x32_f16: (neg_a, A, neg_b, B, c_mod, C, reuse_a, reuse_b)
  return __builtin_amdgcn_wmma_f32_16x16x32_f16(false, a, false, b, (short)0, c, false, false);
}

// A fragment (16x32 f16) from row-major [m][k] half array, row stride DIM.
__device__ __forceinline__ h16 load_a(const _Float16* base, int m0, int k0, int lane) {
  int m  = m0 + (lane & 15);
  int kA = k0 + ((lane >> 4) << 3);
  const _Float16* p = base + m * DIM + kA;
  h8 lo = *(const h8*)p;
  h8 hi = *(const h8*)(p + 16);
  h16 r;
#pragma unroll
  for (int j = 0; j < 8; ++j) { r[j] = lo[j]; r[j + 8] = hi[j]; }
  return r;
}

// B fragment (32x16 f16) from transposed [n][k] half array, row stride ldk.
__device__ __forceinline__ h16 load_bT(const _Float16* baseT, int ldk, int n0, int k0, int lane) {
  const _Float16* p = baseT + (n0 + (lane & 15)) * ldk + k0 + ((lane >> 4) << 3);
  h8 lo = *(const h8*)p;
  h8 hi = *(const h8*)(p + 16);
  h16 r;
#pragma unroll
  for (int j = 0; j < 8; ++j) { r[j] = lo[j]; r[j + 8] = hi[j]; }
  return r;
}

// One GCN layer: Agg = Adj^T @ H (rows >= zrow forced to 0), then
// H = relu(H @ Ws + Agg @ Wn + b), updating Hh and HhT.
// aAdj: per-wave precomputed Adj^T A-fragments (invariant across steps).
__device__ __forceinline__ void gnn_layer(_Float16* Hh, _Float16* HhT, _Float16* Agg,
                                          const h16 aAdj[4],
                                          const _Float16* WsT, const _Float16* WnT,
                                          const float* bias, int zrow, int tid) {
  const int lane = tid & 31;
  const int m0   = (tid >> 5) << 4;   // wave owns 16-row block
  // ---- Phase 1: Agg = Adj^T @ H  (A = adjacency frags, B = HhT) ----
  {
    f8 acc[8] = {};
#pragma unroll
    for (int kb = 0; kb < 4; ++kb) {
      h16 bf[8];
#pragma unroll
      for (int nt = 0; nt < 8; ++nt) bf[nt] = load_bT(HhT, DIM, nt * 16, kb * 32, lane);
#pragma unroll
      for (int nt = 0; nt < 8; ++nt) acc[nt] = wmma16(aAdj[kb], bf[nt], acc[nt]);
    }
#pragma unroll
    for (int nt = 0; nt < 8; ++nt) {
      int n = nt * 16 + (lane & 15);
#pragma unroll
      for (int r = 0; r < 8; ++r) {
        int m = m0 + ((lane >> 4) << 3) + r;
        float v = (m >= zrow) ? 0.0f : acc[nt][r];
        Agg[m * DIM + n] = (_Float16)v;
      }
    }
  }
  __syncthreads();
  // ---- Phase 2: H = relu(H @ Ws + Agg @ Wn + b) ----
  {
    f8 acc[8] = {};
#pragma unroll
    for (int kb = 0; kb < 4; ++kb) {
      h16 aH = load_a(Hh,  m0, kb * 32, lane);
      h16 aG = load_a(Agg, m0, kb * 32, lane);
      h16 bS[8];
#pragma unroll
      for (int nt = 0; nt < 8; ++nt) bS[nt] = load_bT(WsT, DIM, nt * 16, kb * 32, lane);
#pragma unroll
      for (int nt = 0; nt < 8; ++nt) acc[nt] = wmma16(aH, bS[nt], acc[nt]);
      h16 bN[8];
#pragma unroll
      for (int nt = 0; nt < 8; ++nt) bN[nt] = load_bT(WnT, DIM, nt * 16, kb * 32, lane);
#pragma unroll
      for (int nt = 0; nt < 8; ++nt) acc[nt] = wmma16(aG, bN[nt], acc[nt]);
    }
#pragma unroll
    for (int nt = 0; nt < 8; ++nt) {
      int n = nt * 16 + (lane & 15);
      float bv = bias[n];
#pragma unroll
      for (int r = 0; r < 8; ++r) {
        int m = m0 + ((lane >> 4) << 3) + r;
        float v = acc[nt][r] + bv;
        v = v > 0.0f ? v : 0.0f;
        _Float16 hv = (_Float16)v;
        Hh[m * DIM + n]  = hv;
        HhT[n * DIM + m] = hv;
      }
    }
  }
  __syncthreads();
}

__global__ void __launch_bounds__(256)
decoder_kernel(const float* __restrict__ z,   const float* __restrict__ pos_emb,
               const float* __restrict__ Ws1, const float* __restrict__ Wn1,
               const float* __restrict__ b1,  const float* __restrict__ Ws2,
               const float* __restrict__ Wn2, const float* __restrict__ b2,
               const float* __restrict__ mW1, const float* __restrict__ mb1,
               const float* __restrict__ mW2, const float* __restrict__ mb2,
               const float* __restrict__ Wo,  const float* __restrict__ bo,
               const int* __restrict__ targets, float* __restrict__ out) {
  extern __shared__ char smem[];
  _Float16* Hh    = (_Float16*)(smem + OFF_HH);
  _Float16* HhT   = (_Float16*)(smem + OFF_HHT);
  _Float16* Agg   = (_Float16*)(smem + OFF_AGG);
  _Float16* Ws1T  = (_Float16*)(smem + OFF_WS1T);
  _Float16* Wn1T  = (_Float16*)(smem + OFF_WN1T);
  _Float16* Ws2T  = (_Float16*)(smem + OFF_WS2T);
  _Float16* Wn2T  = (_Float16*)(smem + OFF_WN2T);
  _Float16* mW1HT = (_Float16*)(smem + OFF_MW1HT);
  unsigned* adjb  = (unsigned*)(smem + OFF_ADJB);
  unsigned* ancb  = (unsigned*)(smem + OFF_ANCB);
  unsigned char* carr = (unsigned char*)(smem + OFF_CARR);
  float* u     = (float*)(smem + OFF_U);
  float* vz    = (float*)(smem + OFF_VZ);
  float* w2    = (float*)(smem + OFF_W2);
  float* htv   = (float*)(smem + OFF_HT);
  float* petv  = (float*)(smem + OFF_PET);
  float* logit = (float*)(smem + OFF_LOGIT);
  float* bias1 = (float*)(smem + OFF_B1);
  float* bias2 = (float*)(smem + OFF_B2);
  float* red   = (float*)(smem + OFF_RED);

  const int tid  = threadIdx.x;
  const int lane = tid & 31;
  const int wid  = tid >> 5;

  // ---------------- setup ----------------
  for (int i = tid; i < NODES * DIM; i += 256) { Hh[i] = (_Float16)0.0f; HhT[i] = (_Float16)0.0f; }
  for (int i = tid; i < DIM * DIM; i += 256) {
    int n = i >> 7, k = i & 127;
    Ws1T[i] = (_Float16)Ws1[k * DIM + n];
    Wn1T[i] = (_Float16)Wn1[k * DIM + n];
    Ws2T[i] = (_Float16)Ws2[k * DIM + n];
    Wn2T[i] = (_Float16)Wn2[k * DIM + n];
  }
  for (int i = tid; i < MHID * DIM; i += 256) {
    int n = i >> 7, k = i & 127;
    mW1HT[i] = (_Float16)mW1[(DIM + k) * MHID + n];   // H-slab of mW1, transposed
  }
  for (int i = tid; i < NODES * 4; i += 256) {
    int d = i >> 2, w = i & 3;
    unsigned bits = 0u;
    for (int b = 0; b < 32; ++b) bits |= (targets[d * NODES + w * 32 + b] ? 1u : 0u) << b;
    adjb[i] = bits;   // Adj^T row bitsets == T rows
    ancb[i] = 0u;
  }
  {
    int j = tid;                      // 256 threads == MHID
    float s = mb1[j];
    for (int k = 0; k < DIM; ++k) s += z[k] * mW1[(2 * DIM + k) * MHID + j];
    vz[j] = s;                        // mb1 + z @ mW1[z-slab]
    w2[j] = mW2[j];
  }
  if (tid < DIM) {
    bias1[tid] = b1[tid];
    bias2[tid] = b2[tid];
    _Float16 hz = (_Float16)z[tid];   // H0[0] = z
    Hh[0 * DIM + tid] = hz;
    HhT[tid * DIM + 0] = hz;
  }
  if (tid == 0) red[8] = 0.0f;        // total log-prob accumulator
  __syncthreads();

  // ancestor transitive closure (bitsets), wave 0, sequential in t
  if (wid == 0) {
    for (int t = 1; t < NODES; ++t) {
      unsigned aw[4] = {0u, 0u, 0u, 0u};
#pragma unroll
      for (int base = 0; base < NODES; base += 32) {
        int i = base + lane;
        unsigned bit = (adjb[t * 4 + (base >> 5)] >> lane) & 1u;  // T[t][i]
        if (bit && (i < t)) {
#pragma unroll
          for (int w = 0; w < 4; ++w) aw[w] |= ancb[i * 4 + w];
          aw[base >> 5] |= (1u << lane);
        }
      }
#pragma unroll
      for (int w = 0; w < 4; ++w) {
        unsigned v = aw[w];
        v |= (unsigned)__shfl_xor((int)v, 16);
        v |= (unsigned)__shfl_xor((int)v, 8);
        v |= (unsigned)__shfl_xor((int)v, 4);
        v |= (unsigned)__shfl_xor((int)v, 2);
        v |= (unsigned)__shfl_xor((int)v, 1);
        aw[w] = v;
      }
      if (lane == 0) {
#pragma unroll
        for (int w = 0; w < 4; ++w) ancb[t * 4 + w] = aw[w];
      }
    }
  }
  __syncthreads();
  if (tid < NODES) {                   // S[t][i] = 2^-c ;  c = sum_{j<t} anc[j][i]
    int i = tid, cc = 0;
    carr[i] = 0;
    for (int t = 1; t < NODES; ++t) {
      cc += (int)((ancb[(t - 1) * 4 + (i >> 5)] >> (i & 31)) & 1u);
      carr[t * NODES + i] = (unsigned char)cc;
    }
  }
  __syncthreads();

  // Precompute this wave's Adj^T A-fragments once (invariant across all steps).
  h16 aAdj[4];
  {
    const int m0 = wid << 4;
#pragma unroll
    for (int kb = 0; kb < 4; ++kb) {
      int m  = m0 + (lane & 15);
      int kA = kb * 32 + ((lane >> 4) << 3);
      unsigned w0 = adjb[m * 4 + (kA >> 5)] >> (kA & 31);
      int k2 = kA + 16;
      unsigned w1 = adjb[m * 4 + (k2 >> 5)] >> (k2 & 31);
      h16 a;
#pragma unroll
      for (int j = 0; j < 8; ++j) {
        a[j]     = ((w0 >> j) & 1u) ? (_Float16)1.0f : (_Float16)0.0f;
        a[j + 8] = ((w1 >> j) & 1u) ? (_Float16)1.0f : (_Float16)0.0f;
      }
      aAdj[kb] = a;
    }
  }

  // ---------------- sequential decode loop ----------------
#pragma unroll 1
  for (int t = 1; t < NODES; ++t) {
    if (tid < DIM) {                   // H[t] = z + pos_emb[t]
      float pe = pos_emb[t * DIM + tid];
      float hv = z[tid] + pe;
      _Float16 hh = (_Float16)hv;
      Hh[t * DIM + tid]  = hh;
      HhT[tid * DIM + t] = hh;
      petv[tid] = pe;
    }
    __syncthreads();

    gnn_layer(Hh, HhT, Agg, aAdj, Ws1T, Wn1T, bias1, t, tid);
    gnn_layer(Hh, HhT, Agg, aAdj, Ws2T, Wn2T, bias2, t, tid);

    if (tid < DIM) htv[tid] = (float)Hh[t * DIM + tid];  // h_t (post-GNN)
    __syncthreads();

    {                                   // u = vz + h_t@W_ht + pe_t@W_pe
      int j = tid;
      float s = vz[j];
      for (int k = 0; k < DIM; ++k) s += htv[k] * mW1[k * MHID + j];
      for (int k = 0; k < DIM; ++k) s += petv[k] * mW1[(3 * DIM + k) * MHID + j];
      u[j] = s;
    }
    __syncthreads();

    {                                   // logits = relu(H@W_H + u) @ w2 + mb2
      const int m0 = wid << 4;
      h16 aH[4];
#pragma unroll
      for (int kb = 0; kb < 4; ++kb) aH[kb] = load_a(Hh, m0, kb * 32, lane);
      float rowsum[8] = {0, 0, 0, 0, 0, 0, 0, 0};
#pragma unroll
      for (int half = 0; half < 2; ++half) {
        f8 acc[8] = {};
#pragma unroll
        for (int kb = 0; kb < 4; ++kb) {
          h16 bf[8];
#pragma unroll
          for (int j = 0; j < 8; ++j)
            bf[j] = load_bT(mW1HT, DIM, (half * 8 + j) * 16, kb * 32, lane);
#pragma unroll
          for (int j = 0; j < 8; ++j) acc[j] = wmma16(aH[kb], bf[j], acc[j]);
        }
#pragma unroll
        for (int j = 0; j < 8; ++j) {
          int n = (half * 8 + j) * 16 + (lane & 15);
          float uv = u[n], wv = w2[n];
#pragma unroll
          for (int r = 0; r < 8; ++r) {
            float v = acc[j][r] + uv;
            v = v > 0.0f ? v : 0.0f;
            rowsum[r] += v * wv;
          }
        }
      }
      float mb2v = mb2[0];
#pragma unroll
      for (int r = 0; r < 8; ++r) {
        float s = rowsum[r];
        s += __shfl_xor(s, 1);
        s += __shfl_xor(s, 2);
        s += __shfl_xor(s, 4);
        s += __shfl_xor(s, 8);
        if ((lane & 15) == 0) logit[m0 + ((lane >> 4) << 3) + r] = s + mb2v;
      }
    }
    __syncthreads();

    {                                   // masked Bernoulli log-prob
      float part = 0.0f;
      if (tid < t) {
        int i = tid;
        float l   = logit[i];
        float sig = 1.0f / (1.0f + __expf(-l));
        float p   = ldexpf(sig, -(int)carr[t * NODES + i]);   // * S[t][i] = 2^-c
        float gt  = (float)targets[t * NODES + i];
        part = gt * __logf(p + EPSV) + (1.0f - gt) * __logf(1.0f - p + EPSV);
      }
      part += __shfl_xor(part, 1);
      part += __shfl_xor(part, 2);
      part += __shfl_xor(part, 4);
      part += __shfl_xor(part, 8);
      part += __shfl_xor(part, 16);
      if (lane == 0) red[wid] = part;
      __syncthreads();
      if (tid == 0) {
        float s = 0.0f;
        for (int w = 0; w < 8; ++w) s += red[w];
        red[8] += s;
      }
      __syncthreads();
    }
  }

  // final full-graph GNN pass (no masking)
  gnn_layer(Hh, HhT, Agg, aAdj, Ws1T, Wn1T, bias1, NODES, tid);
  gnn_layer(Hh, HhT, Agg, aAdj, Ws2T, Wn2T, bias2, NODES, tid);

  // outputs: L (targets as float), X_out = H@Wo + bo, total_logprob
  for (int idx = tid; idx < NODES * NODES; idx += 256)
    out[idx] = (float)targets[idx];
  {
    int i  = tid >> 1;
    int ob = (tid & 1) * 8;
#pragma unroll
    for (int o = 0; o < 8; ++o) {
      float s = bo[ob + o];
      for (int k = 0; k < DIM; ++k) s += (float)Hh[i * DIM + k] * Wo[k * DOUT + ob + o];
      out[NODES * NODES + i * DOUT + ob + o] = s;
    }
  }
  if (tid == 0) out[NODES * NODES + NODES * DOUT] = red[8];
}

extern "C" void kernel_launch(void* const* d_in, const int* in_sizes, int n_in,
                              void* d_out, int out_size, void* d_ws, size_t ws_size,
                              hipStream_t stream) {
  (void)in_sizes; (void)n_in; (void)d_ws; (void)ws_size; (void)out_size;
  hipFuncSetAttribute((const void*)decoder_kernel,
                      hipFuncAttributeMaxDynamicSharedMemorySize, SMEM_BYTES);
  decoder_kernel<<<1, 256, SMEM_BYTES, stream>>>(
      (const float*)d_in[0],  (const float*)d_in[1],  (const float*)d_in[2],
      (const float*)d_in[3],  (const float*)d_in[4],  (const float*)d_in[5],
      (const float*)d_in[6],  (const float*)d_in[7],  (const float*)d_in[8],
      (const float*)d_in[9],  (const float*)d_in[10], (const float*)d_in[11],
      (const float*)d_in[12], (const float*)d_in[13], (const int*)d_in[14],
      (float*)d_out);
}
